// DynamicSparseAttention_24283745091778
// MI455X (gfx1250) — matrix operand order repaired and checked by reading, not compiled, not run on hardware
//
#include <hip/hip_runtime.h>
#include <hip/hip_bf16.h>

// ---------------------------------------------------------------------------
// DynamicSparseAttention for gfx1250 (MI455X, wave32, WMMA).
// B=8, N=128, D=512, H=8, HD=64, D2=256, D4=128, E=5, K_TOP=89.
// All heavy GEMMs use v_wmma_f32_16x16x32_f16.
// ---------------------------------------------------------------------------

#define Bb 8
#define Nn 128
#define Dd 512
#define Hh 8
#define HD 64
#define D2 256
#define D4 128
#define Ee 5
#define KTOP 89

typedef __attribute__((ext_vector_type(16))) _Float16 v16h;
typedef __attribute__((ext_vector_type(8)))  _Float16 v8h;
typedef __attribute__((ext_vector_type(8)))  float    v8f;
typedef _Float16 half_t;

// ---------------------------------------------------------------------------
// WMMA fragment helpers (CDNA5 16x16x32 f16 layouts, ISA 7.12.2)
// A (16x32): lane m = lane&15; kbase = (lane>>4)*8; halves map to
//            k = kbase + h (h<8) and k = 16 + kbase + (h-8).
// B (32x16) loaded from row-major [N][K] (i.e. computing A @ B_rm^T):
//            n = lane&15; same k mapping as A (two v8h loads).
// C/D (16x16 f32): VGPR r holds M = r + 8*(lane>>4), N = lane&15.
// ---------------------------------------------------------------------------
__device__ __forceinline__ v16h wmma_load_a(const half_t* p, int ld, int lane, int k0) {
  const int m  = lane & 15;
  const int kb = (lane >> 4) * 8;
  const half_t* q = p + (size_t)m * ld + k0;
  v8h lo = *(const v8h*)(q + kb);
  v8h hi = *(const v8h*)(q + 16 + kb);
  v16h r;
#pragma unroll
  for (int h = 0; h < 8; ++h) { r[h] = lo[h]; r[h + 8] = hi[h]; }
  return r;
}

__device__ __forceinline__ v16h wmma_load_b(const half_t* p, int ld, int lane, int k0) {
  const int n = lane & 15;
  const int kb = (lane >> 4) * 8;
  const half_t* q = p + (size_t)n * ld + k0;
  v8h lo = *(const v8h*)(q + kb);
  v8h hi = *(const v8h*)(q + 16 + kb);
  v16h r;
#pragma unroll
  for (int h = 0; h < 8; ++h) { r[h] = lo[h]; r[h + 8] = hi[h]; }
  return r;
}

__device__ __forceinline__ v8f wmma_f16(v16h a, v16h b, v8f c) {
  return __builtin_amdgcn_wmma_f32_16x16x32_f16(false, a, false, b, (short)0, c,
                                                false, false);
}

// ---------------------------------------------------------------------------
// K1: f32 -> f16 (sub)matrix pack, row-major preserved.
// ---------------------------------------------------------------------------
__global__ __launch_bounds__(256)
void pack_f16_kernel(const float* __restrict__ src, half_t* __restrict__ dst,
                     int R, int C, int c0, int cc) {
  int idx = blockIdx.x * blockDim.x + threadIdx.x;
  if (idx >= R * cc) return;
  int r = idx / cc, c = idx % cc;
  dst[idx] = (half_t)src[(size_t)r * C + c0 + c];
}

// ---------------------------------------------------------------------------
// K2: generic C[M][N] = A[M][K] @ B_rm[N][K]^T + bias, one wave per 16x16 tile.
// ---------------------------------------------------------------------------
__global__ __launch_bounds__(256)
void gemm_rt_kernel(const half_t* __restrict__ A, const half_t* __restrict__ Bm,
                    const float* __restrict__ bias, float* __restrict__ C,
                    int M, int N, int K) {
  const int lane = threadIdx.x & 31;
  const int wave = blockIdx.x * (blockDim.x >> 5) + (threadIdx.x >> 5);
  const int ntiles = N >> 4;
  if (wave >= (M >> 4) * ntiles) return;        // wave-uniform: EXEC stays all-1
  const int mt = wave / ntiles, nt = wave % ntiles;
  const half_t* Ab = A + (size_t)mt * 16 * K;
  const half_t* Bb_ = Bm + (size_t)nt * 16 * K;
  v8f acc = {};
  for (int k0 = 0; k0 < K; k0 += 32) {
    __builtin_prefetch(Bb_ + k0 + 128, 0, 1);   // global_prefetch on streamed B
    v16h a = wmma_load_a(Ab, K, lane, k0);
    v16h b = wmma_load_b(Bb_, K, lane, k0);
    acc = wmma_f16(a, b, acc);
  }
  const int n = nt * 16 + (lane & 15);
  const float bv = bias ? bias[n] : 0.0f;
  const int mbase = mt * 16 + ((lane >> 4) * 8);
#pragma unroll
  for (int r = 0; r < 8; ++r) C[(size_t)(mbase + r) * N + n] = acc[r] + bv;
}

// ---------------------------------------------------------------------------
// K3: QKV GEMM with scatter into q/k/v f16 buffers [B,H,N,HD]; q pre-scaled.
//     M=1024 rows of x, N=1536 = {q|k|v} x 512, K=512.
// ---------------------------------------------------------------------------
__global__ __launch_bounds__(256)
void gemm_qkv_kernel(const half_t* __restrict__ A, const half_t* __restrict__ Bm,
                     half_t* __restrict__ qh, half_t* __restrict__ kh,
                     half_t* __restrict__ vh) {
  const int K = Dd, N = 3 * Dd, M = Bb * Nn;
  const int lane = threadIdx.x & 31;
  const int wave = blockIdx.x * (blockDim.x >> 5) + (threadIdx.x >> 5);
  const int ntiles = N >> 4;
  if (wave >= (M >> 4) * ntiles) return;
  const int mt = wave / ntiles, nt = wave % ntiles;
  const half_t* Ab = A + (size_t)mt * 16 * K;
  const half_t* Bp = Bm + (size_t)nt * 16 * K;
  v8f acc = {};
  for (int k0 = 0; k0 < K; k0 += 32) {
    v16h a = wmma_load_a(Ab, K, lane, k0);
    v16h b = wmma_load_b(Bp, K, lane, k0);
    acc = wmma_f16(a, b, acc);
  }
  const int n = nt * 16 + (lane & 15);
  const int s = n >> 9;                 // 0:q 1:k 2:v
  const int hh = (n >> 6) & 7;
  const int dd = n & 63;
  half_t* dst = (s == 0) ? qh : (s == 1) ? kh : vh;
  const float sc = (s == 0) ? 0.125f : 1.0f;   // HD^-0.5 = 1/8
  const int mbase = mt * 16 + ((lane >> 4) * 8);
#pragma unroll
  for (int r = 0; r < 8; ++r) {
    const int row = mbase + r;
    const int b = row >> 7, nd = row & 127;
    dst[((((size_t)b * Hh + hh) * Nn) + nd) * HD + dd] = (half_t)(acc[r] * sc);
  }
}

// ---------------------------------------------------------------------------
// K4: attention scores S[b,h] = (q*scale) @ k^T, per-(b,h) block, 8 waves.
// ---------------------------------------------------------------------------
__global__ __launch_bounds__(256)
void attn_score_kernel(const half_t* __restrict__ qh, const half_t* __restrict__ kh,
                       float* __restrict__ attn) {
  const int bh = blockIdx.x;
  const int lane = threadIdx.x & 31;
  const int mt = threadIdx.x >> 5;           // 8 waves = 8 M-tiles
  const half_t* qb = qh + (size_t)bh * Nn * HD + (size_t)mt * 16 * HD;
  for (int nt = 0; nt < 8; ++nt) {
    const half_t* kb_ = kh + (size_t)bh * Nn * HD + (size_t)nt * 16 * HD;
    v8f acc = {};
#pragma unroll
    for (int k0 = 0; k0 < HD; k0 += 32) {
      v16h a = wmma_load_a(qb, HD, lane, k0);
      v16h b = wmma_load_b(kb_, HD, lane, k0);
      acc = wmma_f16(a, b, acc);
    }
    const int n = nt * 16 + (lane & 15);
    const int mbase = mt * 16 + ((lane >> 4) * 8);
#pragma unroll
    for (int r = 0; r < 8; ++r)
      attn[((size_t)bh * Nn + mbase + r) * Nn + n] = acc[r];
  }
}

// ---------------------------------------------------------------------------
// K5: chemical distances: argmax bond -> weight -> Floyd-Warshall in LDS,
//     write cd = clip(dist,0,3)/3.  One workgroup per batch.
// ---------------------------------------------------------------------------
__global__ __launch_bounds__(256)
void chem_dist_kernel(const float* __restrict__ ef, const unsigned char* __restrict__ mask,
                      float* __restrict__ cd) {
  __shared__ float Dm[Nn][Nn];                 // 64 KB
  const int b = blockIdx.x;
  const int tid = threadIdx.x;
  const float table[5] = {__builtin_inff(), 1.0f, 0.8f, 0.6f, 0.9f};
  for (int idx = tid; idx < Nn * Nn; idx += 256) {
    const int i = idx >> 7, j = idx & 127;
    const float* e = ef + (((size_t)b * Nn + i) * Nn + j) * Ee;
    int am = 0; float bv = e[0];
#pragma unroll
    for (int t = 1; t < Ee; ++t) if (e[t] > bv) { bv = e[t]; am = t; }
    const bool pv = (mask[b * Nn + i] != 0) && (mask[b * Nn + j] != 0);
    float d = pv ? table[am] : __builtin_inff();
    if (i == j && pv) d = 0.0f;
    Dm[i][j] = d;
  }
  __syncthreads();
  for (int kk = 0; kk < Nn; ++kk) {
    for (int idx = tid; idx < Nn * Nn; idx += 256) {
      const int i = idx >> 7, j = idx & 127;
      const float via = Dm[i][kk] + Dm[kk][j];
      if (via < Dm[i][j]) Dm[i][j] = via;
    }
    __syncthreads();
  }
  for (int idx = tid; idx < Nn * Nn; idx += 256) {
    const int i = idx >> 7, j = idx & 127;
    cd[((size_t)b * Nn + i) * Nn + j] =
        fminf(fmaxf(Dm[i][j], 0.0f), 3.0f) * (1.0f / 3.0f);
  }
}

// ---------------------------------------------------------------------------
// K6: fused per-pair sparsity MLP. One block per (b,i); 4 waves x 2 passes,
//     each wave owns a 16-row j-tile.  edge_feat is generated directly in
//     WMMA A-fragment layout (registers, no LDS staging / no barrier for it).
//     Chain: edge_proj(256->128) -> h1(385->256,relu) -> h2(256->128,relu)
//            -> probs(128->8,sigmoid) -> sp[b,h,i,j].
// ---------------------------------------------------------------------------
__global__ __launch_bounds__(128)
void sparsity_mlp_kernel(const float* __restrict__ np, const float* __restrict__ cd,
                         const half_t* __restrict__ wep, const float* __restrict__ bep,
                         const half_t* __restrict__ w1a, const half_t* __restrict__ w1b,
                         const float* __restrict__ w1full, const float* __restrict__ b1,
                         const half_t* __restrict__ w2h, const float* __restrict__ b2,
                         const float* __restrict__ w3, const float* __restrict__ b3,
                         const unsigned char* __restrict__ mask,
                         float* __restrict__ sp) {
  __shared__ __align__(32) half_t stage[4 * 6144];  // 48 KB: 12 KB per wave
  const int b = blockIdx.x >> 7;
  const int i = blockIdx.x & 127;
  const int wave = threadIdx.x >> 5, lane = threadIdx.x & 31;
  half_t* ep_s = &stage[wave * 6144];         // 16x128  (reused for h2)
  half_t* h1_s = &stage[wave * 6144 + 2048];  // 16x256
  half_t* h2_s = ep_s;
  const float* np_i = np + ((size_t)b * Nn + i) * D2;
  const bool mi = mask[b * Nn + i] != 0;
  const int m = lane & 15, kb = (lane >> 4) * 8, mhi = (lane >> 4) * 8;

  for (int pass = 0; pass < 2; ++pass) {
    const int jt = (wave + pass * 4) * 16;    // j-tile base

    // --- 1. edge_feat tile straight into A-frag registers -----------------
    v16h aef[8];
    const float* np_j = np + ((size_t)b * Nn + jt + m) * D2;
#pragma unroll
    for (int kt = 0; kt < 8; ++kt) {
#pragma unroll
      for (int h = 0; h < 8; ++h) {
        const int k1 = kt * 32 + kb + h;
        const int k2 = kt * 32 + 16 + kb + h;
        aef[kt][h]     = (half_t)tanhf(np_i[k1] + np_j[k1]);
        aef[kt][h + 8] = (half_t)tanhf(np_i[k2] + np_j[k2]);
      }
    }

    // --- 2. edge_proj = edge_feat @ wep^T + bep  (linear) -----------------
    for (int nt = 0; nt < 8; ++nt) {
      v8f acc = {};
#pragma unroll
      for (int kt = 0; kt < 8; ++kt)
        acc = wmma_f16(aef[kt], wmma_load_b(wep + nt * 16 * D2, D2, lane, kt * 32), acc);
      const int n = nt * 16 + (lane & 15);
      const float bb = bep[n];
#pragma unroll
      for (int r = 0; r < 8; ++r) ep_s[(mhi + r) * D4 + n] = (half_t)(acc[r] + bb);
    }
    __syncthreads();

    // preload edge_proj A-frags (frees ep_s region for h2 later)
    v16h aep[4];
#pragma unroll
    for (int kt = 0; kt < 4; ++kt) aep[kt] = wmma_load_a(ep_s, D4, lane, kt * 32);
    __syncthreads();

    // --- 3. h1 = relu([ef|ep|cd] @ w1^T + b1) -----------------------------
    float cdn[8];
    const float* cdrow = cd + ((size_t)b * Nn + i) * Nn;
#pragma unroll
    for (int r = 0; r < 8; ++r) cdn[r] = cdrow[jt + mhi + r];  // already /3-normed
    for (int nt = 0; nt < 16; ++nt) {
      v8f acc = {};
#pragma unroll
      for (int kt = 0; kt < 8; ++kt)
        acc = wmma_f16(aef[kt], wmma_load_b(w1a + nt * 16 * D2, D2, lane, kt * 32), acc);
#pragma unroll
      for (int kt = 0; kt < 4; ++kt)
        acc = wmma_f16(aep[kt], wmma_load_b(w1b + nt * 16 * D4, D4, lane, kt * 32), acc);
      const int n = nt * 16 + (lane & 15);
      const float wc = w1full[(size_t)n * 385 + 384];
      const float bb = b1[n];
#pragma unroll
      for (int r = 0; r < 8; ++r)
        h1_s[(mhi + r) * D2 + n] = (half_t)fmaxf(acc[r] + bb + cdn[r] * wc, 0.0f);
    }
    __syncthreads();

    // --- 4. h2 = relu(h1 @ w2^T + b2) ------------------------------------
    for (int nt = 0; nt < 8; ++nt) {
      v8f acc = {};
#pragma unroll
      for (int kt = 0; kt < 8; ++kt) {
        v16h a = wmma_load_a(h1_s, D2, lane, kt * 32);
        acc = wmma_f16(a, wmma_load_b(w2h + nt * 16 * D2, D2, lane, kt * 32), acc);
      }
      const int n = nt * 16 + (lane & 15);
      const float bb = b2[n];
#pragma unroll
      for (int r = 0; r < 8; ++r)
        h2_s[(mhi + r) * D4 + n] = (half_t)fmaxf(acc[r] + bb, 0.0f);
    }
    __syncthreads();

    // --- 5. probs = sigmoid(h2 @ w3^T + b3); write gate sp ----------------
    for (int t = lane; t < 16 * Hh; t += 32) {
      const int jj = t >> 3, hh = t & 7;
      float s = b3[hh];
      for (int c = 0; c < D4; ++c) s += (float)h2_s[jj * D4 + c] * w3[hh * D4 + c];
      const float p = 1.0f / (1.0f + __expf(-s));
      const int j = jt + jj;
      const bool pv = mi && (mask[b * Nn + j] != 0);
      const float spv = (i == j) ? 1.0f : (pv ? p : 0.0f);
      sp[((((size_t)b * Hh + hh) * Nn) + i) * Nn + j] = spv;
    }
    __syncthreads();
  }
}

// ---------------------------------------------------------------------------
// K7: per-(b,h): masked = attn*sp; keep top-89 per row (rank via wave32
//     shuffles), scatter into zeros, softmax, then AW @ V via WMMA.
// ---------------------------------------------------------------------------
__global__ __launch_bounds__(256)
void topk_av_kernel(const float* __restrict__ attn, const float* __restrict__ sp,
                    const half_t* __restrict__ vh, half_t* __restrict__ outh) {
  __shared__ __align__(32) half_t awh[Nn * Nn];   // 32 KB
  const int bh = blockIdx.x;
  const int wave = threadIdx.x >> 5, lane = threadIdx.x & 31;

  for (int q = wave; q < Nn; q += 8) {
    const float* a = attn + ((size_t)bh * Nn + q) * Nn;
    const float* s = sp + ((size_t)bh * Nn + q) * Nn;
    float v[4];
#pragma unroll
    for (int t = 0; t < 4; ++t) { const int c = lane + 32 * t; v[t] = a[c] * s[c]; }
    // rank of each element among the 128 (strict greater-than count)
    int cnt[4] = {0, 0, 0, 0};
    for (int src = 0; src < 32; ++src) {
#pragma unroll
      for (int t = 0; t < 4; ++t) {
        const float bv = __shfl(v[t], src, 32);
#pragma unroll
        for (int u = 0; u < 4; ++u) cnt[u] += (bv > v[u]) ? 1 : 0;
      }
    }
    float sv[4];
#pragma unroll
    for (int u = 0; u < 4; ++u) sv[u] = (cnt[u] < KTOP) ? v[u] : 0.0f;
    // softmax over the sparse row (dropped entries are exact zeros)
    float mx = fmaxf(fmaxf(sv[0], sv[1]), fmaxf(sv[2], sv[3]));
    for (int off = 16; off > 0; off >>= 1) mx = fmaxf(mx, __shfl_xor(mx, off, 32));
    float e[4], sum = 0.0f;
#pragma unroll
    for (int u = 0; u < 4; ++u) { e[u] = __expf(sv[u] - mx); sum += e[u]; }
    for (int off = 16; off > 0; off >>= 1) sum += __shfl_xor(sum, off, 32);
    const float inv = 1.0f / sum;
#pragma unroll
    for (int t = 0; t < 4; ++t) awh[q * Nn + lane + 32 * t] = (half_t)(e[t] * inv);
  }
  __syncthreads();

  // out[b, q, h*64+d] = AW @ V ; wave = M-tile, loop over 4 N-tiles (HD=64)
  const int mt = wave;
  const half_t* aw = awh + mt * 16 * Nn;
  const int b = bh >> 3, hh = bh & 7;
  for (int nt = 0; nt < 4; ++nt) {
    const half_t* vb = vh + (size_t)bh * Nn * HD + (size_t)nt * 16 * HD;
    v8f acc = {};
#pragma unroll
    for (int k0 = 0; k0 < Nn; k0 += 32) {
      v16h afr = wmma_load_a(aw, Nn, lane, k0);
      // V rows are keys (k) and cols are dims (n) -> B from row-major [k][n]:
      // need B(k,n)=V[k][n]; our loader expects row-major [N][K], so gather.
      v16h bfr;
      const int n = lane & 15;
      const int kbase = k0 + (lane >> 4) * 8;
#pragma unroll
      for (int h = 0; h < 16; ++h) bfr[h] = vb[(size_t)(kbase + (h & 7) + (h >> 3) * 16 - k0) * HD + n + (size_t)k0 * HD];
      acc = wmma_f16(afr, bfr, acc);
    }
    const int n = nt * 16 + (lane & 15);
    const int mbase = mt * 16 + ((lane >> 4) * 8);
#pragma unroll
    for (int r = 0; r < 8; ++r)
      outh[((size_t)b * Nn + mbase + r) * Dd + hh * HD + n] = (half_t)acc[r];
  }
}

// ---------------------------------------------------------------------------
// Host launcher
// ---------------------------------------------------------------------------
extern "C" void kernel_launch(void* const* d_in, const int* in_sizes, int n_in,
                              void* d_out, int out_size, void* d_ws, size_t ws_size,
                              hipStream_t stream) {
  (void)in_sizes; (void)n_in; (void)out_size; (void)ws_size;
  const float* x      = (const float*)d_in[0];
  const unsigned char* mask = (const unsigned char*)d_in[1];
  const float* ef     = (const float*)d_in[2];
  const float* w_qkv  = (const float*)d_in[3];
  const float* w_proj = (const float*)d_in[4];
  const float* b_proj = (const float*)d_in[5];
  const float* w_np   = (const float*)d_in[6];
  const float* b_np   = (const float*)d_in[7];
  const float* w_ep   = (const float*)d_in[8];
  const float* b_ep   = (const float*)d_in[9];
  const float* w1     = (const float*)d_in[10];
  const float* b1     = (const float*)d_in[11];
  const float* w2     = (const float*)d_in[12];
  const float* b2     = (const float*)d_in[13];
  const float* w3     = (const float*)d_in[14];
  const float* b3     = (const float*)d_in[15];
  float* out = (float*)d_out;

  char* ws = (char*)d_ws;
  size_t off = 0;
  auto alloc = [&](size_t bytes) -> void* {
    void* p = ws + off;
    off = (off + bytes + 255) & ~(size_t)255;
    return p;
  };
  const int M = Bb * Nn;                       // 1024
  half_t* xh     = (half_t*)alloc((size_t)M * Dd * 2);
  half_t* wqkvh  = (half_t*)alloc((size_t)3 * Dd * Dd * 2);
  half_t* wnph   = (half_t*)alloc((size_t)D2 * Dd * 2);
  half_t* wprojh = (half_t*)alloc((size_t)Dd * Dd * 2);
  half_t* weph   = (half_t*)alloc((size_t)D4 * D2 * 2);
  half_t* w1ah   = (half_t*)alloc((size_t)D2 * D2 * 2);
  half_t* w1bh   = (half_t*)alloc((size_t)D2 * D4 * 2);
  half_t* w2h    = (half_t*)alloc((size_t)D4 * D2 * 2);
  half_t* qh     = (half_t*)alloc((size_t)Bb * Hh * Nn * HD * 2);
  half_t* kh     = (half_t*)alloc((size_t)Bb * Hh * Nn * HD * 2);
  half_t* vh     = (half_t*)alloc((size_t)Bb * Hh * Nn * HD * 2);
  float*  npf    = (float*)alloc((size_t)M * D2 * 4);
  float*  attn   = (float*)alloc((size_t)Bb * Hh * Nn * Nn * 4);
  float*  spf    = (float*)alloc((size_t)Bb * Hh * Nn * Nn * 4);
  float*  cdf    = (float*)alloc((size_t)Bb * Nn * Nn * 4);
  half_t* outh   = (half_t*)alloc((size_t)M * Dd * 2);

  auto PACK = [&](const float* s, half_t* d, int R, int C, int c0, int cc) {
    int n = R * cc;
    pack_f16_kernel<<<(n + 255) / 256, 256, 0, stream>>>(s, d, R, C, c0, cc);
  };
  // weight + activation packs (f32 -> f16, row-major)
  PACK(x,      xh,     M,        Dd,  0,   Dd);
  PACK(w_qkv,  wqkvh,  3 * Dd,   Dd,  0,   Dd);
  PACK(w_np,   wnph,   D2,       Dd,  0,   Dd);
  PACK(w_proj, wprojh, Dd,       Dd,  0,   Dd);
  PACK(w_ep,   weph,   D4,       Dd,  0,   D2);   // only first D2 cols matter
  PACK(w1,     w1ah,   D2,       385, 0,   D2);
  PACK(w1,     w1bh,   D2,       385, D2,  D4);
  PACK(w2,     w2h,    D4,       D2,  0,   D2);

  // QKV:   x @ w_qkv^T  -> q/k/v f16 (q pre-scaled)
  {
    int waves = (M / 16) * ((3 * Dd) / 16);
    gemm_qkv_kernel<<<(waves + 7) / 8, 256, 0, stream>>>(xh, wqkvh, qh, kh, vh);
  }
  // node_proj: x @ w_np^T + b_np -> f32
  {
    int waves = (M / 16) * (D2 / 16);
    gemm_rt_kernel<<<(waves + 7) / 8, 256, 0, stream>>>(xh, wnph, b_np, npf, M, D2, Dd);
  }
  // attention scores
  attn_score_kernel<<<Bb * Hh, 256, 0, stream>>>(qh, kh, attn);
  // chemical distances (Floyd-Warshall in LDS) -> cd normalized
  chem_dist_kernel<<<Bb, 256, 0, stream>>>(ef, mask, cdf);
  // fused per-pair sparsity MLP -> sp gates
  sparsity_mlp_kernel<<<Bb * Nn, 128, 0, stream>>>(npf, cdf, weph, b_ep, w1ah, w1bh,
                                                   w1, b1, w2h, b2, w3, b3, mask, spf);
  // top-k + softmax + AW @ V
  topk_av_kernel<<<Bb * Hh, 256, 0, stream>>>(attn, spf, vh, outh);
  // final projection: out = attn_out @ w_proj^T + b_proj
  {
    int waves = (M / 16) * (Dd / 16);
    gemm_rt_kernel<<<(waves + 7) / 8, 256, 0, stream>>>(outh, wprojh, b_proj, out,
                                                        M, Dd, Dd);
  }
}
